// TransformerBlock_17643725652488
// MI455X (gfx1250) — matrix operand order repaired and checked by reading, not compiled, not run on hardware
//
#include <hip/hip_runtime.h>

typedef unsigned short u16;
typedef __attribute__((ext_vector_type(16))) __bf16 v16bf;
typedef __attribute__((ext_vector_type(8)))  float  v8f;

struct U4x2 { uint4 a, b; };

__device__ __forceinline__ u16 f2bf(float f) {
    unsigned int u = __builtin_bit_cast(unsigned int, f);
    u += 0x7fffu + ((u >> 16) & 1u);   // round-to-nearest-even
    return (u16)(u >> 16);
}

// ---- CDNA5 async global->LDS copy (16 B per lane), tracked by ASYNCcnt -----
__device__ __forceinline__ void async_cp16(const u16* g, u16* l) {
    unsigned lds = (unsigned)(size_t)l;                 // flat addr[31:0] == LDS offset
    unsigned long long ga = (unsigned long long)(size_t)g;
    asm volatile("global_load_async_to_lds_b128 %0, %1, off"
                 :: "v"(lds), "v"(ga) : "memory");
}
__device__ __forceinline__ void async_wait0() {
    asm volatile("s_wait_asynccnt 0x0" ::: "memory");
}

// ---------------- cast + transpose weights: src[K,N] f32 -> dst[N,K] bf16 ----
__global__ __launch_bounds__(256) void k_castT(const float* __restrict__ src,
                                               u16* __restrict__ dst,
                                               int K, int N) {
    long i = (long)blockIdx.x * 256 + threadIdx.x;
    if (i < (long)N * K) {
        int n = (int)(i / K);
        int k = (int)(i % K);
        dst[i] = f2bf(src[(long)k * N + n]);
    }
}

// ---------------- LayerNorm over D=1024, fp32 in -> bf16 out -----------------
__global__ __launch_bounds__(256) void k_ln(const float* __restrict__ x,
                                            const float* __restrict__ g,
                                            const float* __restrict__ b,
                                            u16* __restrict__ out, int D) {
    __shared__ float s1[256], s2[256];
    long row = blockIdx.x;
    const float* xr = x + row * (long)D;
    int t = threadIdx.x;
    float v[4], sum = 0.f, sq = 0.f;
    #pragma unroll
    for (int i = 0; i < 4; i++) {
        v[i] = xr[t + 256 * i];
        sum += v[i]; sq += v[i] * v[i];
    }
    s1[t] = sum; s2[t] = sq; __syncthreads();
    for (int s = 128; s > 0; s >>= 1) {
        if (t < s) { s1[t] += s1[t + s]; s2[t] += s2[t + s]; }
        __syncthreads();
    }
    float mean = s1[0] * (1.0f / 1024.0f);
    float var  = s2[0] * (1.0f / 1024.0f) - mean * mean;
    float rstd = rsqrtf(var + 1e-5f);
    u16* orow = out + row * (long)D;
    #pragma unroll
    for (int i = 0; i < 4; i++) {
        int c = t + 256 * i;
        orow[c] = f2bf((v[i] - mean) * rstd * g[c] + b[c]);
    }
}

// ---------------- row softmax over 2048, fp32 in -> bf16 out -----------------
__global__ __launch_bounds__(256) void k_softmax(const float* __restrict__ S,
                                                 u16* __restrict__ P) {
    __shared__ float red[256];
    long row = blockIdx.x;
    const float* sr = S + row * 2048L;
    int t = threadIdx.x;
    float v[8], m = -3.4e38f;
    #pragma unroll
    for (int i = 0; i < 8; i++) { v[i] = sr[t + 256 * i]; m = fmaxf(m, v[i]); }
    red[t] = m; __syncthreads();
    for (int s = 128; s > 0; s >>= 1) {
        if (t < s) red[t] = fmaxf(red[t], red[t + s]);
        __syncthreads();
    }
    m = red[0]; __syncthreads();
    float sum = 0.f;
    #pragma unroll
    for (int i = 0; i < 8; i++) { v[i] = __expf(v[i] - m); sum += v[i]; }
    red[t] = sum; __syncthreads();
    for (int s = 128; s > 0; s >>= 1) {
        if (t < s) red[t] += red[t + s];
        __syncthreads();
    }
    float inv = 1.0f / red[0];
    u16* pr = P + row * 2048L;
    #pragma unroll
    for (int i = 0; i < 8; i++) pr[t + 256 * i] = f2bf(v[i] * inv);
}

// ---------------- bf16 WMMA GEMM, NT form: C[m,n] = sum_k A[m,k]*Bt[n,k] -----
// Block tile 128x128, K-step 32, double-buffered async LDS staging.
// 256 threads = 8 waves; wave tile 64(M)x32(N) = 4x2 accum fragments.
template<int GELU, int OBF16, int OTRANS, int RES>
__global__ __launch_bounds__(256) void k_gemm(
    const u16* __restrict__ A,  int lda, long aB,
    const u16* __restrict__ Bt, int ldb, long bB,
    const float* __restrict__ bias,
    const float* __restrict__ res, int ldr, long rB,
    void* __restrict__ Cv, int ldc, long cB,
    int K, float alpha)
{
    __shared__ __align__(16) u16 As[2][128 * 32];
    __shared__ __align__(16) u16 Bs[2][128 * 32];

    const int bz = blockIdx.z;
    A  += (long)bz * aB;
    Bt += (long)bz * bB;

    const int m0 = blockIdx.y * 128, n0 = blockIdx.x * 128;
    const int tid = threadIdx.x;
    const int lane = tid & 31, wave = tid >> 5;
    const int wm = (wave & 1) * 64, wn = (wave >> 1) * 32;
    const int l15 = lane & 15, lh = lane >> 4;

    v8f acc[4][2];
    #pragma unroll
    for (int i = 0; i < 4; i++)
        #pragma unroll
        for (int j = 0; j < 2; j++)
            #pragma unroll
            for (int e = 0; e < 8; e++) acc[i][j][e] = 0.0f;

    const int lr = tid >> 1;          // tile row handled by this thread (0..127)
    const int lc = (tid & 1) << 4;    // half-element offset (0 or 16)
    const u16* gA = A  + (long)(m0 + lr) * lda + lc;
    const u16* gB = Bt + (long)(n0 + lr) * ldb + lc;

    const int nk = K >> 5;            // number of 32-wide k-tiles

    // prologue: issue async loads for tile 0 into buffer 0
    async_cp16(gA,     &As[0][lr * 32 + lc]);
    async_cp16(gA + 8, &As[0][lr * 32 + lc + 8]);
    async_cp16(gB,     &Bs[0][lr * 32 + lc]);
    async_cp16(gB + 8, &Bs[0][lr * 32 + lc + 8]);

    for (int kk = 0; kk < nk; kk++) {
        const int cur = kk & 1;
        async_wait0();                // my tile-kk copies have landed
        __syncthreads();              // everyone's landed; prev-buffer reads done

        if (kk + 1 < nk) {            // overlap: issue tile kk+1 into other buffer
            const u16* nA = gA + ((kk + 1) << 5);
            const u16* nB = gB + ((kk + 1) << 5);
            const int nxt = cur ^ 1;
            async_cp16(nA,     &As[nxt][lr * 32 + lc]);
            async_cp16(nA + 8, &As[nxt][lr * 32 + lc + 8]);
            async_cp16(nB,     &Bs[nxt][lr * 32 + lc]);
            async_cp16(nB + 8, &Bs[nxt][lr * 32 + lc + 8]);
        }

        // A frag: lane holds row M, halves K = {kh*8..+7, 16+kh*8..+7}
        v16bf af[4], bfg[2];
        #pragma unroll
        for (int mf = 0; mf < 4; mf++) {
            int m = wm + mf * 16 + l15;
            const uint4* p = (const uint4*)((const char*)As[cur] + m * 64 + lh * 16);
            U4x2 u; u.a = p[0]; u.b = p[2];   // +0 and +32 bytes
            af[mf] = __builtin_bit_cast(v16bf, u);
        }
        // B frag: lane holds col N, halves K = {kh*8..+7, 16+kh*8..+7}
        #pragma unroll
        for (int nf = 0; nf < 2; nf++) {
            int n = wn + nf * 16 + l15;
            const uint4* p = (const uint4*)((const char*)Bs[cur] + n * 64 + lh * 16);
            U4x2 u; u.a = p[0]; u.b = p[2];
            bfg[nf] = __builtin_bit_cast(v16bf, u);
        }
        #pragma unroll
        for (int mf = 0; mf < 4; mf++)
            #pragma unroll
            for (int nf = 0; nf < 2; nf++)
                acc[mf][nf] = __builtin_amdgcn_wmma_f32_16x16x32_bf16(
                    false, af[mf], false, bfg[nf],
                    (short)0, acc[mf][nf], false, false);
    }

    // epilogue: alpha, bias, residual, GELU, store (f32 / bf16 / bf16-transposed)
    const float* rp = res + (RES ? (long)bz * rB : 0);
    float* C32 = (float*)Cv;
    u16*   C16 = (u16*)Cv;
    const long cbo = (long)bz * cB;
    #pragma unroll
    for (int mf = 0; mf < 4; mf++) {
        #pragma unroll
        for (int nf = 0; nf < 2; nf++) {
            int n = n0 + wn + nf * 16 + l15;
            float bv = bias ? bias[n] : 0.0f;
            #pragma unroll
            for (int vi = 0; vi < 8; vi++) {
                int m = m0 + wm + mf * 16 + lh * 8 + vi;
                float v = acc[mf][nf][vi] * alpha + bv;
                if (RES)  v += rp[(long)m * ldr + n];
                if (GELU) v = 0.5f * v * (1.0f + erff(v * 0.70710678118f));
                if (OTRANS)      C16[cbo + (long)n * ldc + m] = f2bf(v);
                else if (OBF16)  C16[cbo + (long)m * ldc + n] = f2bf(v);
                else             C32[cbo + (long)m * ldc + n] = v;
            }
        }
    }
}

extern "C" void kernel_launch(void* const* d_in, const int* in_sizes, int n_in,
                              void* d_out, int out_size, void* d_ws, size_t ws_size,
                              hipStream_t stream) {
    const float* x     = (const float*)d_in[0];
    const float* ln1_g = (const float*)d_in[1];
    const float* ln1_b = (const float*)d_in[2];
    const float* W_q   = (const float*)d_in[3];
    const float* b_q   = (const float*)d_in[4];
    const float* W_k   = (const float*)d_in[5];
    const float* b_k   = (const float*)d_in[6];
    const float* W_v   = (const float*)d_in[7];
    const float* b_v   = (const float*)d_in[8];
    const float* ln2_g = (const float*)d_in[9];
    const float* ln2_b = (const float*)d_in[10];
    const float* W_in  = (const float*)d_in[11];
    const float* b_in  = (const float*)d_in[12];
    const float* W_out = (const float*)d_in[13];
    const float* b_out = (const float*)d_in[14];

    const int Bn = 4, T = 2048, D = 1024, DH = 1024, DFF = 4096, BT = Bn * T;
    const size_t MB = 1024 * 1024;
    char* W = (char*)d_ws;
    u16*   wq   = (u16*)  (W +   0 * MB);   //  2 MB  [DH, D]
    u16*   wk   = (u16*)  (W +   2 * MB);   //  2 MB
    u16*   wv   = (u16*)  (W +   4 * MB);   //  2 MB
    u16*   win  = (u16*)  (W +   6 * MB);   //  8 MB  [DFF, D]
    u16*   wout = (u16*)  (W +  14 * MB);   //  8 MB  [D, DFF]
    u16*   h    = (u16*)  (W +  22 * MB);   // 16 MB  [BT, D]  (reused for h2)
    u16*   Qb   = (u16*)  (W +  38 * MB);   // 16 MB  [BT, DH]
    u16*   Kb   = (u16*)  (W +  54 * MB);   // 16 MB  [BT, DH]
    u16*   Vt   = (u16*)  (W +  70 * MB);   // 16 MB  [DH, BT] (V transposed)
    float* x1   = (float*)(W +  86 * MB);   // 32 MB  [BT, D]
    float* S    = (float*)(W + 118 * MB);   // 64 MB  [B, T, T] (reused as ff)
    u16*   ff   = (u16*)  (W + 118 * MB);   // 64 MB  [BT, DFF]
    u16*   P    = (u16*)  (W + 182 * MB);   // 32 MB  [B, T, T] bf16

    // 1) cast + transpose weights to bf16 [N,K]
    k_castT<<<(D * DH   + 255) / 256, 256, 0, stream>>>(W_q,   wq,   D,   DH);
    k_castT<<<(D * DH   + 255) / 256, 256, 0, stream>>>(W_k,   wk,   D,   DH);
    k_castT<<<(D * DH   + 255) / 256, 256, 0, stream>>>(W_v,   wv,   D,   DH);
    k_castT<<<(D * DFF  + 255) / 256, 256, 0, stream>>>(W_in,  win,  D,   DFF);
    k_castT<<<(DFF * D  + 255) / 256, 256, 0, stream>>>(W_out, wout, DFF, D);

    // 2) LN1
    k_ln<<<BT, 256, 0, stream>>>(x, ln1_g, ln1_b, h, D);

    // 3) Q, K, V projections (V written transposed for NT attention GEMM)
    dim3 gqkv(DH / 128, BT / 128, 1);
    k_gemm<0,1,0,0><<<gqkv, 256, 0, stream>>>(h, D, 0, wq, D, 0, b_q,
        nullptr, 0, 0, Qb, DH, 0, D, 1.0f);
    k_gemm<0,1,0,0><<<gqkv, 256, 0, stream>>>(h, D, 0, wk, D, 0, b_k,
        nullptr, 0, 0, Kb, DH, 0, D, 1.0f);
    k_gemm<0,1,1,0><<<gqkv, 256, 0, stream>>>(h, D, 0, wv, D, 0, b_v,
        nullptr, 0, 0, Vt, BT, 0, D, 1.0f);          // ldc = BT (transposed)

    // 4) scores S = Q K^T * 1/sqrt(DH), batched over B
    dim3 gs(T / 128, T / 128, Bn);
    k_gemm<0,0,0,0><<<gs, 256, 0, stream>>>(Qb, DH, (long)T * DH,
        Kb, DH, (long)T * DH, nullptr, nullptr, 0, 0,
        S, T, (long)T * T, DH, 1.0f / 32.0f);

    // 5) softmax rows -> P (bf16)
    k_softmax<<<Bn * T, 256, 0, stream>>>(S, P);

    // 6) attn_out = P @ V + x  -> x1 (fp32)
    dim3 ga(DH / 128, T / 128, Bn);
    k_gemm<0,0,0,1><<<ga, 256, 0, stream>>>(P, T, (long)T * T,
        Vt, BT, (long)T, nullptr, x, D, (long)T * D,
        x1, D, (long)T * D, T, 1.0f);

    // 7) LN2 (reuse h)
    k_ln<<<BT, 256, 0, stream>>>(x1, ln2_g, ln2_b, h, D);

    // 8) ff = gelu(h2 @ W_in + b_in)  (bf16, reuses S region)
    dim3 gf(DFF / 128, BT / 128, 1);
    k_gemm<1,1,0,0><<<gf, 256, 0, stream>>>(h, D, 0, win, D, 0, b_in,
        nullptr, 0, 0, ff, DFF, 0, D, 1.0f);

    // 9) out = ff @ W_out + b_out + x1 -> d_out (fp32)
    dim3 go(D / 128, BT / 128, 1);
    k_gemm<0,0,0,1><<<go, 256, 0, stream>>>(ff, DFF, 0, wout, DFF, 0, b_out,
        x1, D, 0, (float*)d_out, D, 0, DFF, 1.0f);
}
